// Attention_69295002353902
// MI455X (gfx1250) — matrix operand order repaired and checked by reading, not compiled, not run on hardware
//
#include <hip/hip_runtime.h>
#include <cstdint>
#include <cstddef>

// ---------------------------------------------------------------------------
// Types for CDNA5 WMMA (wave32, 16x16x32 bf16 -> f32)
// ---------------------------------------------------------------------------
typedef __attribute__((ext_vector_type(16))) __bf16    v16bf;
typedef __attribute__((ext_vector_type(8)))  float     v8f;
typedef __attribute__((ext_vector_type(8)))  uint32_t  u32x8;

union FragAB { u32x8 u; v16bf h; };

__device__ __forceinline__ v8f wmma_bf16(const FragAB& a, const FragAB& b, v8f c) {
    return __builtin_amdgcn_wmma_f32_16x16x32_bf16(
        false, a.h, false, b.h, (short)0, c, false, false);
}

__device__ __forceinline__ uint16_t f32_bf16(float f) {
    union { float f; uint32_t u; } c{f};
    uint32_t u = c.u;
    uint32_t r = (u + 0x7FFFu + ((u >> 16) & 1u)) >> 16;  // RNE
    return (uint16_t)r;
}
__device__ __forceinline__ float bf16_f32(uint16_t h) {
    union { uint32_t u; float f; } c{(uint32_t)h << 16};
    return c.f;
}

// ---- gfx1250 async global->LDS path (signature confirmed by compiler):
// (int vector_size(16) AS1* gsrc, AS3* ldst, int offset, int cpol)
#if __has_builtin(__builtin_amdgcn_global_load_async_to_lds_b128) && \
    __has_builtin(__builtin_amdgcn_s_wait_asynccnt)
#define HAVE_ASYNC_LDS 1
typedef int async_v4i __attribute__((vector_size(16)));
typedef __attribute__((address_space(1))) async_v4i* gas_ptr_t;
typedef __attribute__((address_space(3))) async_v4i* las_ptr_t;
#endif

// ---------------------------------------------------------------------------
// Problem constants
// ---------------------------------------------------------------------------
#define BATCH 4
#define SEQ   2048
#define HID   2048
#define NH    16
#define HD    128
#define NQKV  6144

// ---------------------------------------------------------------------------
// Kernel 1: f32 -> bf16 cast
// ---------------------------------------------------------------------------
__global__ __launch_bounds__(256) void cast_bf16_kernel(const float* __restrict__ src,
                                                        uint16_t* __restrict__ dst,
                                                        size_t n) {
    size_t i = (size_t)blockIdx.x * 256 + threadIdx.x;
    if (i < n) dst[i] = f32_bf16(src[i]);
}

// ---------------------------------------------------------------------------
// Kernel 2/5: tiled bf16 GEMM, C = A(MxK)*B(KxN), f32 accumulate.
// 128x128 block tile, BK=32, 8 waves (2x4), wave = 64x32 = 4x2 WMMA tiles.
// A tile streamed via async global->LDS DMA (ASYNCcnt); B tile via register
// prefetch + transpose into [n][k]. One barrier per K-step.
// MODE 0: scatter q/k/v bf16 [b][h][s][d];  MODE 1: f32 row-major out.
// ---------------------------------------------------------------------------
#define LDA 34   // 17 dwords/row: coprime with 64 banks

template <int MODE>
__global__ __launch_bounds__(256)
void gemm_bf16_kernel(const uint16_t* __restrict__ A,
                      const uint16_t* __restrict__ Bm,
                      int M, int N, int K,
                      uint16_t* __restrict__ qb,
                      uint16_t* __restrict__ kb,
                      uint16_t* __restrict__ vb,
                      float* __restrict__ out) {
    __shared__ uint16_t As[2][128 * LDA];   // [m][k]
    __shared__ uint16_t Bs[2][128 * LDA];   // [n][k] (transposed at store)

    const int tid  = threadIdx.x;
    const int lane = tid & 31;
    const int wid  = tid >> 5;
    const int wm   = wid & 1;
    const int wn   = wid >> 1;
    const int hlf  = lane >> 4;
    const int l16  = lane & 15;
    const int bm0  = blockIdx.y * 128;
    const int bn0  = blockIdx.x * 128;

    int rA[2], cA[2], kB[2], nB[2];
#pragma unroll
    for (int i = 0; i < 2; ++i) {
        int c = tid + i * 256;
        rA[i] = c >> 2;          cA[i] = (c & 3) * 8;
        kB[i] = c >> 4;          nB[i] = (c & 15) * 8;
    }

    v8f acc[4][2];
    const v8f vzero = 0.0f;
#pragma unroll
    for (int mi = 0; mi < 4; ++mi)
#pragma unroll
        for (int ni = 0; ni < 2; ++ni) acc[mi][ni] = vzero;

    // ---- prologue prefetch (K-tile 0) ----
    uint4 rb[2];
#if defined(HAVE_ASYNC_LDS)
#pragma unroll
    for (int i = 0; i < 2; ++i)
        __builtin_amdgcn_global_load_async_to_lds_b128(
            (gas_ptr_t)(A + (size_t)(bm0 + rA[i]) * K + cA[i]),
            (las_ptr_t)&As[0][rA[i] * LDA + cA[i]], 0, 0);
#else
    uint4 ra[2];
#pragma unroll
    for (int i = 0; i < 2; ++i)
        ra[i] = *(const uint4*)(A + (size_t)(bm0 + rA[i]) * K + cA[i]);
#endif
#pragma unroll
    for (int i = 0; i < 2; ++i)
        rb[i] = *(const uint4*)(Bm + (size_t)kB[i] * N + bn0 + nB[i]);

    int buf = 0;
    for (int k0 = 0; k0 < K; k0 += 32) {
        // ---- commit B (transposed) and A into LDS[buf] ----
#pragma unroll
        for (int i = 0; i < 2; ++i) {
            uint32_t w[4] = {rb[i].x, rb[i].y, rb[i].z, rb[i].w};
#pragma unroll
            for (int e = 0; e < 8; ++e)
                Bs[buf][(nB[i] + e) * LDA + kB[i]] =
                    (uint16_t)(w[e >> 1] >> ((e & 1) * 16));
#if !defined(HAVE_ASYNC_LDS)
            uint32_t* dst = (uint32_t*)&As[buf][rA[i] * LDA + cA[i]];
            dst[0] = ra[i].x; dst[1] = ra[i].y; dst[2] = ra[i].z; dst[3] = ra[i].w;
#endif
        }
#if defined(HAVE_ASYNC_LDS)
        __builtin_amdgcn_s_wait_asynccnt(0);
#endif
        __syncthreads();

        // ---- prefetch next K-tile (overlaps with WMMAs below) ----
        if (k0 + 32 < K) {
            int kn = k0 + 32;
#if defined(HAVE_ASYNC_LDS)
#pragma unroll
            for (int i = 0; i < 2; ++i)
                __builtin_amdgcn_global_load_async_to_lds_b128(
                    (gas_ptr_t)(A + (size_t)(bm0 + rA[i]) * K + kn + cA[i]),
                    (las_ptr_t)&As[buf ^ 1][rA[i] * LDA + cA[i]], 0, 0);
#else
#pragma unroll
            for (int i = 0; i < 2; ++i)
                ra[i] = *(const uint4*)(A + (size_t)(bm0 + rA[i]) * K + kn + cA[i]);
#endif
#pragma unroll
            for (int i = 0; i < 2; ++i)
                rb[i] = *(const uint4*)(Bm + (size_t)(kn + kB[i]) * N + bn0 + nB[i]);
        }

        // ---- fragments + 8 WMMAs ----
        FragAB af[4], bfr[2];
#pragma unroll
        for (int mi = 0; mi < 4; ++mi) {
            const uint32_t* row =
                (const uint32_t*)&As[buf][(wm * 64 + mi * 16 + l16) * LDA];
#pragma unroll
            for (int j = 0; j < 8; ++j) {
                int kk = (j < 4 ? 2 * j : 16 + 2 * (j - 4)) + hlf * 8;
                af[mi].u[j] = row[kk >> 1];
            }
        }
#pragma unroll
        for (int ni = 0; ni < 2; ++ni) {
            const uint32_t* row =
                (const uint32_t*)&Bs[buf][(wn * 32 + ni * 16 + l16) * LDA];
#pragma unroll
            for (int j = 0; j < 8; ++j)
                bfr[ni].u[j] = row[((j < 4 ? 2 * j : 16 + 2 * (j - 4)) + hlf * 8) >> 1];
        }
#pragma unroll
        for (int mi = 0; mi < 4; ++mi)
#pragma unroll
            for (int ni = 0; ni < 2; ++ni)
                acc[mi][ni] = wmma_bf16(af[mi], bfr[ni], acc[mi][ni]);
        buf ^= 1;
    }

    // ---- epilogue ----
#pragma unroll
    for (int mi = 0; mi < 4; ++mi)
#pragma unroll
        for (int ni = 0; ni < 2; ++ni)
#pragma unroll
            for (int r = 0; r < 8; ++r) {
                int grow = bm0 + wm * 64 + mi * 16 + r + hlf * 8;
                int gcol = bn0 + wn * 32 + ni * 16 + l16;
                float val = acc[mi][ni][r];
                if (MODE == 0) {
                    int b   = grow >> 11;
                    int s   = grow & (SEQ - 1);
                    int sec = gcol >> 11;
                    int rem = gcol & 2047;
                    int h   = rem >> 7;
                    int d   = rem & (HD - 1);
                    size_t idx = (((size_t)b * NH + h) * SEQ + s) * HD + d;
                    uint16_t* dst = (sec == 0) ? qb : (sec == 1) ? kb : vb;
                    dst[idx] = f32_bf16(val);
                } else {
                    out[(size_t)grow * N + gcol] = val;
                }
            }
}

// ---------------------------------------------------------------------------
// Kernel 3: RoPE in place on q,k
// ---------------------------------------------------------------------------
__global__ __launch_bounds__(256)
void rope_kernel(uint16_t* __restrict__ q, uint16_t* __restrict__ k,
                 const float* __restrict__ cs, const float* __restrict__ sn) {
    size_t p   = (size_t)blockIdx.x * 256 + threadIdx.x;
    int d      = (int)(p & 63);
    size_t row = p >> 6;
    int s      = (int)(row & (SEQ - 1));
    size_t i0  = row * HD + d;
    size_t i1  = i0 + 64;
    float c0 = cs[(size_t)s * HD + d], c1 = cs[(size_t)s * HD + d + 64];
    float s0 = sn[(size_t)s * HD + d], s1 = sn[(size_t)s * HD + d + 64];

    float qa = bf16_f32(q[i0]), qb_ = bf16_f32(q[i1]);
    q[i0] = f32_bf16(qa * c0 - qb_ * s0);
    q[i1] = f32_bf16(qb_ * c1 + qa * s1);
    float ka = bf16_f32(k[i0]), kb_ = bf16_f32(k[i1]);
    k[i0] = f32_bf16(ka * c0 - kb_ * s0);
    k[i1] = f32_bf16(kb_ * c1 + ka * s1);
}

// ---------------------------------------------------------------------------
// Kernel 4: flash attention. Block = (b,h, 128 q-rows), 8 waves x 16 rows.
// Double-buffered K/V tiles; K tile via async global->LDS DMA.
// ---------------------------------------------------------------------------
#define LDK 130
#define LDV 34
#define LDP 34

__global__ __launch_bounds__(256)
void flash_attn_kernel(const uint16_t* __restrict__ q,
                       const uint16_t* __restrict__ k,
                       const uint16_t* __restrict__ v,
                       uint16_t* __restrict__ attn) {
    __shared__ uint16_t Ks[2][32 * LDK];    // [col][d]
    __shared__ uint16_t Vs[2][HD * LDV];    // [d][col] (transposed)
    __shared__ uint16_t Ps[8][16 * LDP];    // per-wave P scratch

    const int bh   = blockIdx.y;
    const int qblk = blockIdx.x;
    const int tid  = threadIdx.x;
    const int lane = tid & 31;
    const int wid  = tid >> 5;
    const int hlf  = lane >> 4;
    const int l16  = lane & 15;
    const size_t base = (size_t)bh * SEQ * HD;
    const int qrow0 = qblk * 128 + wid * 16;
    const float scale = 0.08838834764831845f;   // 1/sqrt(128)

    int kR[2], dC[2];
#pragma unroll
    for (int i = 0; i < 2; ++i) {
        int c = tid + i * 256;
        kR[i] = c >> 4;
        dC[i] = (c & 15) * 8;
    }

    // ---- Q fragments (registers for the whole kernel) ----
    FragAB qf[4];
    {
        const uint16_t* qrow = q + base + (size_t)(qrow0 + l16) * HD;
#pragma unroll
        for (int f = 0; f < 4; ++f)
#pragma unroll
            for (int j = 0; j < 8; ++j) {
                int dd = f * 32 + (j < 4 ? 2 * j : 16 + 2 * (j - 4)) + hlf * 8;
                qf[f].u[j] = *(const uint32_t*)&qrow[dd];
            }
    }

    float mrow[8], lrow[8];
    v8f acc[8];
    const v8f vzero = 0.0f;
#pragma unroll
    for (int r = 0; r < 8; ++r) { mrow[r] = -3.0e38f; lrow[r] = 0.0f; }
#pragma unroll
    for (int nd = 0; nd < 8; ++nd) acc[nd] = vzero;

    // ---- prologue prefetch (tile 0) ----
    uint4 rv[2];
#if defined(HAVE_ASYNC_LDS)
#pragma unroll
    for (int i = 0; i < 2; ++i)
        __builtin_amdgcn_global_load_async_to_lds_b128(
            (gas_ptr_t)(k + base + (size_t)kR[i] * HD + dC[i]),
            (las_ptr_t)&Ks[0][kR[i] * LDK + dC[i]], 0, 0);
#else
    uint4 rk[2];
#pragma unroll
    for (int i = 0; i < 2; ++i)
        rk[i] = *(const uint4*)(k + base + (size_t)kR[i] * HD + dC[i]);
#endif
#pragma unroll
    for (int i = 0; i < 2; ++i)
        rv[i] = *(const uint4*)(v + base + (size_t)kR[i] * HD + dC[i]);

    int buf = 0;
    for (int kb = 0; kb < SEQ; kb += 32) {
        // ---- commit V (transposed) and K into LDS[buf] ----
#pragma unroll
        for (int i = 0; i < 2; ++i) {
            uint32_t w[4] = {rv[i].x, rv[i].y, rv[i].z, rv[i].w};
#pragma unroll
            for (int e = 0; e < 8; ++e)
                Vs[buf][(dC[i] + e) * LDV + kR[i]] =
                    (uint16_t)(w[e >> 1] >> ((e & 1) * 16));
#if !defined(HAVE_ASYNC_LDS)
            uint32_t* dk = (uint32_t*)&Ks[buf][kR[i] * LDK + dC[i]];
            dk[0] = rk[i].x; dk[1] = rk[i].y; dk[2] = rk[i].z; dk[3] = rk[i].w;
#endif
        }
#if defined(HAVE_ASYNC_LDS)
        __builtin_amdgcn_s_wait_asynccnt(0);
#endif
        __syncthreads();

        // ---- prefetch next tile (overlaps with WMMAs below) ----
        if (kb + 32 < SEQ) {
            size_t nbase = base + (size_t)(kb + 32) * HD;
#if defined(HAVE_ASYNC_LDS)
#pragma unroll
            for (int i = 0; i < 2; ++i)
                __builtin_amdgcn_global_load_async_to_lds_b128(
                    (gas_ptr_t)(k + nbase + (size_t)kR[i] * HD + dC[i]),
                    (las_ptr_t)&Ks[buf ^ 1][kR[i] * LDK + dC[i]], 0, 0);
#else
#pragma unroll
            for (int i = 0; i < 2; ++i)
                rk[i] = *(const uint4*)(k + nbase + (size_t)kR[i] * HD + dC[i]);
#endif
#pragma unroll
            for (int i = 0; i < 2; ++i)
                rv[i] = *(const uint4*)(v + nbase + (size_t)kR[i] * HD + dC[i]);
        }

        // ---- scores: Q(16x128) . K^T -> two 16x16 subtiles ----
        v8f sc[2];
        sc[0] = vzero; sc[1] = vzero;
#pragma unroll
        for (int ni = 0; ni < 2; ++ni) {
            const uint32_t* krow =
                (const uint32_t*)&Ks[buf][(ni * 16 + l16) * LDK];
#pragma unroll
            for (int f = 0; f < 4; ++f) {
                FragAB bfr;
#pragma unroll
                for (int j = 0; j < 8; ++j)
                    bfr.u[j] = krow[(f * 32 + (j < 4 ? 2 * j : 16 + 2 * (j - 4)) + hlf * 8) >> 1];
                sc[ni] = wmma_bf16(qf[f], bfr, sc[ni]);
            }
        }

        // ---- online softmax (fully unrolled) ----
        float corr[8];
#pragma unroll
        for (int r = 0; r < 8; ++r) {
            float s0 = sc[0][r] * scale;
            float s1 = sc[1][r] * scale;
            float mx = fmaxf(s0, s1);
#pragma unroll
            for (int off = 1; off < 16; off <<= 1)
                mx = fmaxf(mx, __shfl_xor(mx, off, 32));
            float mn = fmaxf(mrow[r], mx);
            corr[r]  = __expf(mrow[r] - mn);
            mrow[r]  = mn;

            float p0 = __expf(s0 - mn);
            float p1 = __expf(s1 - mn);
            int prow = r + hlf * 8;
            Ps[wid][prow * LDP + l16]      = f32_bf16(p0);
            Ps[wid][prow * LDP + 16 + l16] = f32_bf16(p1);
            float ps = p0 + p1;
#pragma unroll
            for (int off = 1; off < 16; off <<= 1)
                ps += __shfl_xor(ps, off, 32);
            lrow[r] = lrow[r] * corr[r] + ps;
        }
#pragma unroll
        for (int nd = 0; nd < 8; ++nd)
#pragma unroll
            for (int r = 0; r < 8; ++r)
                acc[nd][r] *= corr[r];

        // ---- P as A-fragment (same-wave LDS, in-order) ----
        FragAB pf;
#pragma unroll
        for (int j = 0; j < 8; ++j) {
            int kk = (j < 4 ? 2 * j : 16 + 2 * (j - 4)) + hlf * 8;
            pf.u[j] = *(const uint32_t*)&Ps[wid][l16 * LDP + kk];
        }
        // ---- P(16x32) @ V(32x128) ----
#pragma unroll
        for (int nd = 0; nd < 8; ++nd) {
            const uint32_t* vrow =
                (const uint32_t*)&Vs[buf][(nd * 16 + l16) * LDV];
            FragAB vf;
#pragma unroll
            for (int j = 0; j < 8; ++j)
                vf.u[j] = vrow[((j < 4 ? 2 * j : 16 + 2 * (j - 4)) + hlf * 8) >> 1];
            acc[nd] = wmma_bf16(pf, vf, acc[nd]);
        }
        buf ^= 1;
    }

    // ---- epilogue: O = acc / l ----
    int b = bh >> 4;
    int h = bh & 15;
#pragma unroll
    for (int nd = 0; nd < 8; ++nd)
#pragma unroll
        for (int r = 0; r < 8; ++r) {
            int s   = qrow0 + r + hlf * 8;
            int col = h * HD + nd * 16 + l16;
            float o = acc[nd][r] / lrow[r];
            attn[((size_t)b * SEQ + s) * (NH * HD) + col] = f32_bf16(o);
        }
}

// ---------------------------------------------------------------------------
// Host launcher
// ---------------------------------------------------------------------------
extern "C" void kernel_launch(void* const* d_in, const int* in_sizes, int n_in,
                              void* d_out, int out_size, void* d_ws, size_t ws_size,
                              hipStream_t stream) {
    const float* hs    = (const float*)d_in[0];
    const float* cosp  = (const float*)d_in[1];
    const float* sinp  = (const float*)d_in[2];
    const float* wqkv  = (const float*)d_in[3];
    const float* wo    = (const float*)d_in[4];
    float*       out   = (float*)d_out;

    const size_t n_hs   = (size_t)BATCH * SEQ * HID;
    const size_t n_wqkv = (size_t)HID * NQKV;
    const size_t n_wo   = (size_t)HID * HID;
    const size_t n_qkv1 = (size_t)BATCH * NH * SEQ * HD;

    uint16_t* hs_bf   = (uint16_t*)d_ws;
    uint16_t* wqkv_bf = hs_bf + n_hs;
    uint16_t* wo_bf   = wqkv_bf + n_wqkv;
    uint16_t* qb      = wo_bf + n_wo;
    uint16_t* kb      = qb + n_qkv1;
    uint16_t* vb      = kb + n_qkv1;
    uint16_t* attn    = vb + n_qkv1;

    cast_bf16_kernel<<<(uint32_t)(n_hs / 256),   256, 0, stream>>>(hs,   hs_bf,   n_hs);
    cast_bf16_kernel<<<(uint32_t)(n_wqkv / 256), 256, 0, stream>>>(wqkv, wqkv_bf, n_wqkv);
    cast_bf16_kernel<<<(uint32_t)(n_wo / 256),   256, 0, stream>>>(wo,   wo_bf,   n_wo);

    {
        dim3 grid(NQKV / 128, (BATCH * SEQ) / 128);
        gemm_bf16_kernel<0><<<grid, 256, 0, stream>>>(
            hs_bf, wqkv_bf, BATCH * SEQ, NQKV, HID, qb, kb, vb, nullptr);
    }
    {
        size_t pairs = (size_t)BATCH * NH * SEQ * 64;
        rope_kernel<<<(uint32_t)(pairs / 256), 256, 0, stream>>>(qb, kb, cosp, sinp);
    }
    {
        dim3 grid(SEQ / 128, BATCH * NH);
        flash_attn_kernel<<<grid, 256, 0, stream>>>(qb, kb, vb, attn);
    }
    {
        dim3 grid(HID / 128, (BATCH * SEQ) / 128);
        gemm_bf16_kernel<1><<<grid, 256, 0, stream>>>(
            attn, wo_bf, BATCH * SEQ, HID, HID, nullptr, nullptr, nullptr, out);
    }
}
